// HomeostaticSpikingLayer_48361331753696
// MI455X (gfx1250) — matrix operand order repaired and checked by reading, not compiled, not run on hardware
//
#include <hip/hip_runtime.h>

#define B_N   128
#define T_N   64
#define IN_N  2048
#define OUT_N 2048

#define TARGET_RATE 0.05f
#define TRACE_DECAY 0.95122942450071400910f  // exp(-1/20)
#define ACT_DECAY   0.999f
#define BETA_       0.9f
#define ETA_TH      0.001f
#define MIN_W      -5.0f
#define MAX_W       5.0f
#define MIN_TH      0.5f
#define MAX_TH      2.0f

typedef __attribute__((ext_vector_type(16))) __bf16 v16bf;
typedef __attribute__((ext_vector_type(8)))  __bf16 v8bf;
typedef __attribute__((ext_vector_type(4)))  __bf16 v4bf;
typedef __attribute__((ext_vector_type(8)))  float  v8f;
typedef __attribute__((ext_vector_type(4)))  float  v4f;

// ---------------------------------------------------------------------------
// One-time init: copy W to workspace + produce its bf16 hi/lo split, zero v,
// init per-neuron state vectors.
// ---------------------------------------------------------------------------
__global__ __launch_bounds__(256)
void init_kernel(const float* __restrict__ W, const float* __restrict__ th_in,
                 float* __restrict__ Wf, __bf16* __restrict__ Whi,
                 __bf16* __restrict__ Wlo, float* __restrict__ v,
                 float* __restrict__ post_sum, float* __restrict__ post_tr,
                 float* __restrict__ post_act, float* __restrict__ th) {
  int idx = blockIdx.x * 256 + threadIdx.x;
  if (idx < OUT_N * IN_N) {
    float wv = W[idx];
    Wf[idx] = wv;
    __bf16 h = (__bf16)wv;
    Whi[idx] = h;
    Wlo[idx] = (__bf16)(wv - (float)h);
  }
  if (idx < B_N * OUT_N) v[idx] = 0.0f;
  if (idx < OUT_N) {
    post_sum[idx] = 0.0f;
    post_tr[idx]  = 0.0f;
    post_act[idx] = TARGET_RATE;
    th[idx]       = th_in[idx];
  }
}

// x (0/1 spikes) -> bf16, exact. Done once for all 64 steps.
__global__ __launch_bounds__(256)
void xcvt_kernel(const float* __restrict__ x, __bf16* __restrict__ xbf) {
  int idx = (blockIdx.x * 256 + threadIdx.x) * 4;
  v4f xf = *(const v4f*)&x[idx];
  *(v4bf*)&xbf[idx] = __builtin_convertvector(xf, v4bf);
}

// pre_rate[t][i] = mean over batch of x[:,t,i]
__global__ __launch_bounds__(256)
void reduce_pre_kernel(const float* __restrict__ x, float* __restrict__ preRate) {
  int idx = blockIdx.x * 256 + threadIdx.x;      // t*IN + i
  int t = idx >> 11;
  int i = idx & (IN_N - 1);
  float s = 0.0f;
  for (int b = 0; b < B_N; ++b) s += x[(b * T_N + t) * IN_N + i];
  preRate[idx] = s * (1.0f / B_N);
}

// pre_tr[t][i] scan: tr = decay*tr + pre_rate[t][i]
__global__ __launch_bounds__(256)
void scan_pre_kernel(const float* __restrict__ preRate, float* __restrict__ preTr) {
  int i = blockIdx.x * 256 + threadIdx.x;
  float tr = 0.0f;
  for (int t = 0; t < T_N; ++t) {
    tr = TRACE_DECAY * tr + preRate[t * IN_N + i];
    preTr[t * IN_N + i] = tr;
  }
}

// ---------------------------------------------------------------------------
// K1: v = BETA*v + x_t @ W^T via split-bf16 WMMA. Staging via async LDS-DMA
// (global_load_async_to_lds_b128, ASYNCcnt), quad-buffered LDS, partial
// asynccnt waits for a three-iteration-deep DMA pipeline, peeled drain.
// Block = 256 threads (8 wave32), 32(M) x 64(N) tile, one 16x16 per wave.
// KC=64: two WMMA K-slabs per staged chunk; staging is exactly 5 b128 async
// issues per thread per chunk (uniform, no exec masking).
// ---------------------------------------------------------------------------
#define KC 64
#define LSTRIDE 72                 // bf16 units: 144B rows, conflict-free, 16B-aligned
#define NCHUNK (IN_N / KC)         // 32
#define XSZ_E (32 * LSTRIDE)       // x buffer elements
#define WSZ_E (64 * LSTRIDE)       // W buffer elements

__global__ __launch_bounds__(256)
void gemm_spike_kernel(const __bf16* __restrict__ xbf, const __bf16* __restrict__ Whi,
                       const __bf16* __restrict__ Wlo, float* __restrict__ v,
                       const float* __restrict__ th, float* __restrict__ post_sum,
                       float* __restrict__ out, int t) {
  __shared__ __align__(16) __bf16 xs[4][XSZ_E];
  __shared__ __align__(16) __bf16 wh[4][WSZ_E];
  __shared__ __align__(16) __bf16 wl[4][WSZ_E];

  const int tid  = threadIdx.x;
  const int lane = tid & 31;
  const int wv   = tid >> 5;
  const int mt   = wv & 1;        // M sub-tile (0..1)
  const int nt   = wv >> 1;       // N sub-tile (0..3)
  const int half = lane >> 4;
  const int l16  = lane & 15;
  const int mBlock = blockIdx.y * 32;
  const int nBlock = blockIdx.x * 64;

  // async-DMA lane assignment: 16B (8 bf16) per issue, fully uniform.
  const int xr = tid >> 3;             // 0..31 (x rows)
  const int cc = (tid & 7) * 8;        // 0..56 (cols)
  const int wr0 = tid >> 3;            // W rows 0..31 (a) / 32..63 (b)
  const unsigned xByte0  = (unsigned)((((mBlock + xr) * T_N + t) * IN_N + cc) * 2);
  const unsigned wByte0a = (unsigned)(((nBlock + wr0) * IN_N + cc) * 2);
  const unsigned wByte0b = (unsigned)(((nBlock + wr0 + 32) * IN_N + cc) * 2);

  const unsigned ldsX0  = (unsigned)(size_t)&xs[0][xr * LSTRIDE + cc];
  const unsigned ldsWa0 = (unsigned)(size_t)&wh[0][wr0 * LSTRIDE + cc];
  const unsigned ldsWb0 = (unsigned)(size_t)&wh[0][(wr0 + 32) * LSTRIDE + cc];
  const unsigned ldsLa0 = (unsigned)(size_t)&wl[0][wr0 * LSTRIDE + cc];
  const unsigned ldsLb0 = (unsigned)(size_t)&wl[0][(wr0 + 32) * LSTRIDE + cc];
  const unsigned long long xBase  = (unsigned long long)(size_t)xbf;
  const unsigned long long whBase = (unsigned long long)(size_t)Whi;
  const unsigned long long wlBase = (unsigned long long)(size_t)Wlo;

  auto issue = [&](int buf, int chunk) {
    const unsigned kByte = (unsigned)(chunk * (KC * 2));
    const unsigned xo = (unsigned)(buf * (XSZ_E * 2));
    const unsigned wo = (unsigned)(buf * (WSZ_E * 2));
    asm volatile("global_load_async_to_lds_b128 %0, %1, %2"
                 :: "v"(ldsX0 + xo),  "v"(xByte0 + kByte),  "s"(xBase)  : "memory");
    asm volatile("global_load_async_to_lds_b128 %0, %1, %2"
                 :: "v"(ldsWa0 + wo), "v"(wByte0a + kByte), "s"(whBase) : "memory");
    asm volatile("global_load_async_to_lds_b128 %0, %1, %2"
                 :: "v"(ldsWb0 + wo), "v"(wByte0b + kByte), "s"(whBase) : "memory");
    asm volatile("global_load_async_to_lds_b128 %0, %1, %2"
                 :: "v"(ldsLa0 + wo), "v"(wByte0a + kByte), "s"(wlBase) : "memory");
    asm volatile("global_load_async_to_lds_b128 %0, %1, %2"
                 :: "v"(ldsLb0 + wo), "v"(wByte0b + kByte), "s"(wlBase) : "memory");
  };

  v8f acc = {};
  const int arow = (mt * 16 + l16) * LSTRIDE;
  const int brow = (nt * 16 + l16) * LSTRIDE;

  auto compute = [&](int buf) {
    const __bf16* xsb = &xs[buf][0];
    const __bf16* whb = &wh[buf][0];
    const __bf16* wlb = &wl[buf][0];
#pragma unroll
    for (int s = 0; s < 2; ++s) {
      const int ka = arow + s * 32 + half * 8;
      v8bf a0 = *(const v8bf*)&xsb[ka];
      v8bf a1 = *(const v8bf*)&xsb[ka + 16];
      v16bf a = __builtin_shufflevector(a0, a1, 0, 1, 2, 3, 4, 5, 6, 7,
                                        8, 9, 10, 11, 12, 13, 14, 15);
      const int kb = brow + s * 32 + half * 8;
      v8bf bh0 = *(const v8bf*)&whb[kb];
      v8bf bh1 = *(const v8bf*)&whb[kb + 16];
      v16bf bh = __builtin_shufflevector(bh0, bh1, 0, 1, 2, 3, 4, 5, 6, 7,
                                         8, 9, 10, 11, 12, 13, 14, 15);
      v8bf bl0 = *(const v8bf*)&wlb[kb];
      v8bf bl1 = *(const v8bf*)&wlb[kb + 16];
      v16bf bl = __builtin_shufflevector(bl0, bl1, 0, 1, 2, 3, 4, 5, 6, 7,
                                         8, 9, 10, 11, 12, 13, 14, 15);

      acc = __builtin_amdgcn_wmma_f32_16x16x32_bf16(false, a, false, bh,
                                                    (short)0, acc, false, false);
      acc = __builtin_amdgcn_wmma_f32_16x16x32_bf16(false, a, false, bl,
                                                    (short)0, acc, false, false);
    }
  };

  // Prologue: fill 3 of 4 buffers; require chunk 0 landed (newest 10 may fly).
  issue(0, 0);
  issue(1, 1);
  issue(2, 2);
  asm volatile("s_wait_asynccnt 10" ::: "memory");
  __syncthreads();

  // Steady state: prefetch chunk ck+3, compute chunk ck; the two newest
  // chunks may remain in flight (wait <= 10).
  for (int ck = 0; ck < NCHUNK - 3; ++ck) {
    issue((ck + 3) & 3, ck + 3);
    compute(ck & 3);
    asm volatile("s_wait_asynccnt 10" ::: "memory");
    __syncthreads();
  }
  // Drain: chunks NCHUNK-3, NCHUNK-2, NCHUNK-1.
  compute((NCHUNK - 3) & 3);
  asm volatile("s_wait_asynccnt 5" ::: "memory");
  __syncthreads();
  compute((NCHUNK - 2) & 3);
  asm volatile("s_wait_asynccnt 0" ::: "memory");
  __syncthreads();
  compute((NCHUNK - 1) & 3);

  // Epilogue: LIF membrane + spike + reset; spike column sums.
  // C/D layout: lanes 0-15: N=lane, VGPR r -> M=r; lanes 16-31: M=r+8.
  const int n = nBlock + nt * 16 + l16;
  const float thv = th[n];
  float s = 0.0f;
#pragma unroll
  for (int r = 0; r < 8; ++r) {
    int m = mBlock + mt * 16 + half * 8 + r;
    float vv = BETA_ * v[m * OUT_N + n] + acc[r];
    float spk = (vv >= thv) ? 1.0f : 0.0f;
    v[m * OUT_N + n] = vv * (1.0f - spk);
    out[(m * T_N + t) * OUT_N + n] = spk;
    s += spk;
  }
  atomicAdd(&post_sum[n], s);
}

// ---------------------------------------------------------------------------
// K2: per-neuron state; folds |A+|,|A-|,meta into row factors c1,c2.
// ---------------------------------------------------------------------------
__global__ __launch_bounds__(256)
void vec_update_kernel(const float* __restrict__ Ap, const float* __restrict__ Am,
                       float* __restrict__ post_sum, float* __restrict__ post_tr,
                       float* __restrict__ post_act, float* __restrict__ th,
                       float* __restrict__ c1, float* __restrict__ c2) {
  int n = blockIdx.x * 256 + threadIdx.x;
  float pr = post_sum[n] * (1.0f / B_N);   // post_rate
  post_sum[n] = 0.0f;                      // reset for next step
  float pt = TRACE_DECAY * post_tr[n] + pr;
  post_tr[n] = pt;
  float pa = ACT_DECAY * post_act[n] + (1.0f - ACT_DECAY) * pr;
  post_act[n] = pa;
  float ratio = fminf(fmaxf(pa * (1.0f / TARGET_RATE), 0.5f), 2.0f);
  float meta = 1.0f / ratio;
  c1[n] = pr * fabsf(Ap[0]) * meta;        // LTP row factor
  c2[n] = pt * fabsf(Am[0]) * meta;        // LTD row factor
  float t2 = th[n] * (1.0f + ETA_TH * (pr - TARGET_RATE));
  th[n] = fminf(fmaxf(t2, MIN_TH), MAX_TH);
}

// ---------------------------------------------------------------------------
// K3: W soft-bound STDP update + refresh bf16 hi/lo split for next GEMM.
// ---------------------------------------------------------------------------
__global__ __launch_bounds__(256)
void w_update_kernel(float* __restrict__ Wf, __bf16* __restrict__ Whi,
                     __bf16* __restrict__ Wlo, const float* __restrict__ c1,
                     const float* __restrict__ c2, const float* __restrict__ preTr_t,
                     const float* __restrict__ preRate_t) {
  int idx = (blockIdx.x * 256 + threadIdx.x) * 4;
  int n = idx >> 11;
  int i = idx & (IN_N - 1);
  float a = c1[n], b = c2[n];
  v4f pt = *(const v4f*)&preTr_t[i];
  v4f pr = *(const v4f*)&preRate_t[i];
  v4f w  = *(const v4f*)&Wf[idx];
  v4f dw = a * pt - b * pr;
#pragma unroll
  for (int e = 0; e < 4; ++e) {
    float d = dw[e], wvv = w[e];
    float dwb = (d > 0.0f) ? d * (MAX_W - wvv) * 0.1f : d * (wvv - MIN_W) * 0.1f;
    w[e] = fminf(fmaxf(wvv + dwb, MIN_W), MAX_W);
  }
  *(v4f*)&Wf[idx] = w;
  v4bf h  = __builtin_convertvector(w, v4bf);
  v4f  hf = __builtin_convertvector(h, v4f);
  *(v4bf*)&Whi[idx] = h;
  *(v4bf*)&Wlo[idx] = __builtin_convertvector(w - hf, v4bf);
}

// ---------------------------------------------------------------------------
extern "C" void kernel_launch(void* const* d_in, const int* in_sizes, int n_in,
                              void* d_out, int out_size, void* d_ws, size_t ws_size,
                              hipStream_t stream) {
  const float* x     = (const float*)d_in[0];
  const float* W     = (const float*)d_in[1];
  const float* th_in = (const float*)d_in[2];
  const float* Ap    = (const float*)d_in[3];
  const float* Am    = (const float*)d_in[4];
  float* out = (float*)d_out;

  // workspace carve (~69 MB), 256B-aligned chunks
  char* base = (char*)d_ws;
  auto alloc = [&](size_t bytes) -> char* {
    char* r = base;
    base += (bytes + 255) & ~(size_t)255;
    return r;
  };
  float* Wf       = (float*)alloc((size_t)OUT_N * IN_N * 4);
  float* v        = (float*)alloc((size_t)B_N * OUT_N * 4);
  float* preRate  = (float*)alloc((size_t)T_N * IN_N * 4);
  float* preTr    = (float*)alloc((size_t)T_N * IN_N * 4);
  float* post_sum = (float*)alloc(OUT_N * 4);
  float* post_tr  = (float*)alloc(OUT_N * 4);
  float* post_act = (float*)alloc(OUT_N * 4);
  float* th       = (float*)alloc(OUT_N * 4);
  float* c1       = (float*)alloc(OUT_N * 4);
  float* c2       = (float*)alloc(OUT_N * 4);
  __bf16* xbf     = (__bf16*)alloc((size_t)B_N * T_N * IN_N * 2);
  __bf16* Whi     = (__bf16*)alloc((size_t)OUT_N * IN_N * 2);
  __bf16* Wlo     = (__bf16*)alloc((size_t)OUT_N * IN_N * 2);

  init_kernel<<<(OUT_N * IN_N + 255) / 256, 256, 0, stream>>>(
      W, th_in, Wf, Whi, Wlo, v, post_sum, post_tr, post_act, th);
  xcvt_kernel<<<((size_t)B_N * T_N * IN_N / 4) / 256, 256, 0, stream>>>(x, xbf);
  reduce_pre_kernel<<<(T_N * IN_N) / 256, 256, 0, stream>>>(x, preRate);
  scan_pre_kernel<<<IN_N / 256, 256, 0, stream>>>(preRate, preTr);

  dim3 ggrid(OUT_N / 64, B_N / 32);  // 32 x 4 blocks
  for (int t = 0; t < T_N; ++t) {
    gemm_spike_kernel<<<ggrid, 256, 0, stream>>>(xbf, Whi, Wlo, v, th, post_sum, out, t);
    vec_update_kernel<<<OUT_N / 256, 256, 0, stream>>>(
        Ap, Am, post_sum, post_tr, post_act, th, c1, c2);
    w_update_kernel<<<(OUT_N * IN_N / 4) / 256, 256, 0, stream>>>(
        Wf, Whi, Wlo, c1, c2, preTr + t * IN_N, preRate + t * IN_N);
  }
}
